// GATEncoder_24575802868240
// MI455X (gfx1250) — matrix operand order repaired and checked by reading, not compiled, not run on hardware
//
#include <hip/hip_runtime.h>

// ---------------------------------------------------------------------------
// GAT encoder for MI455X (gfx1250, wave32, WMMA).
// - eft = ef @ We folded into a tiny [EIN x H] projection on raw efeat.
// - GEMMs: 128x128x32 tiles, v_wmma_f32_16x16x32_bf16, 8 WMMA/wave/K-step.
// - Async global->LDS staging (guarded) for the edge-score tables.
// ---------------------------------------------------------------------------

#define IN_   128
#define EIN_  64
#define HID_  256
#define OUT_  256
#define L_    3
#define H_    4
#define D_    64

typedef __attribute__((ext_vector_type(16))) __bf16          bf16x16;
typedef __attribute__((ext_vector_type(8)))  float           f32x8;
typedef __attribute__((ext_vector_type(8)))  unsigned short  u16x8;

union Frag16 {
    bf16x16 bf;
    u16x8   h[2];
};

#if __has_builtin(__builtin_amdgcn_global_load_async_to_lds_b32)
#define HAS_ASYNC_COPY 1
#else
#define HAS_ASYNC_COPY 0
#endif

__device__ __forceinline__ void wait_async0() {
#if __has_builtin(__builtin_amdgcn_s_wait_asynccnt)
    __builtin_amdgcn_s_wait_asynccnt(0);
#elif HAS_ASYNC_COPY
    asm volatile("s_wait_asynccnt 0" ::: "memory");
#endif
}

__device__ __forceinline__ void atomicMaxFloat(float* addr, float val) {
    int* ai = (int*)addr;
    int old = __float_as_int(*addr);
    while (__int_as_float(old) < val) {
        int assumed = old;
        old = atomicCAS(ai, assumed, __float_as_int(val));
        if (old == assumed) break;
    }
}

// ---------------------------------------------------------------------------
// Tiled GEMM: C[M,Nc] = A[M,K] @ B[K,Nc] + bias.  K % 32 == 0, Nc % 128 == 0.
// Block: 256 threads = 8 waves (4x2); 128x128 tile; wave owns 32x64
// (2 A-frags x 4 B-frags -> 8 v_wmma per K-step, 64 per block).
// ---------------------------------------------------------------------------
#define TM 128
#define TN 128
#define TK 32

__global__ __launch_bounds__(256)
void gemm_bias_wmma(const float* __restrict__ A, const float* __restrict__ B,
                    const float* __restrict__ bias, float* __restrict__ C,
                    int M, int K, int Nc) {
    __shared__ __bf16 As[TM][TK];      // 128x32 bf16 row-major   (8 KB)
    __shared__ __bf16 Bs[TN][TK + 8];  // transposed [n][k], padded (10 KB)

    const int tid  = threadIdx.x;
    const int lane = tid & 31;
    const int wave = tid >> 5;   // 0..7
    const int wm   = wave >> 1;  // 0..3 -> rows wm*32 .. wm*32+31
    const int wn   = wave & 1;   // 0..1 -> cols wn*64 .. wn*64+63

    const int bm = blockIdx.y * TM;
    const int bn = blockIdx.x * TN;

    f32x8 acc[2][4];
#pragma unroll
    for (int mi = 0; mi < 2; ++mi)
#pragma unroll
        for (int ni = 0; ni < 4; ++ni)
            acc[mi][ni] = (f32x8){};

    const int row16 = lane & 15;
    const int kh    = lane >> 4;

    for (int k0 = 0; k0 < K; k0 += TK) {
        // ---- A tile: 128x32 f32 -> bf16 (4 float4 per thread) ----
#pragma unroll
        for (int i = 0; i < 4; ++i) {
            int idx = tid + i * 256;          // 0..1023
            int r   = idx >> 3;               // 8 float4 per row
            int c4  = (idx & 7) << 2;
            int gr  = bm + r;
            float4 v = make_float4(0.f, 0.f, 0.f, 0.f);
            if (gr < M) v = *(const float4*)(A + (size_t)gr * K + k0 + c4);
            As[r][c4 + 0] = (__bf16)v.x;
            As[r][c4 + 1] = (__bf16)v.y;
            As[r][c4 + 2] = (__bf16)v.z;
            As[r][c4 + 3] = (__bf16)v.w;
        }
        // ---- B tile: 32x128 f32 -> bf16 transposed ----
#pragma unroll
        for (int i = 0; i < 4; ++i) {
            int idx = tid + i * 256;          // 0..1023 (32 float4 per row)
            int r   = idx >> 5;               // k row
            int c4  = (idx & 31) << 2;        // n col
            float4 v = *(const float4*)(B + (size_t)(k0 + r) * Nc + bn + c4);
            Bs[c4 + 0][r] = (__bf16)v.x;
            Bs[c4 + 1][r] = (__bf16)v.y;
            Bs[c4 + 2][r] = (__bf16)v.z;
            Bs[c4 + 3][r] = (__bf16)v.w;
        }
        // prefetch next K tile (cacheline-granular hints)
        if (k0 + TK < K) {
            int ra = tid & 127;
            if (bm + ra < M)
                __builtin_prefetch(A + (size_t)(bm + ra) * K + (k0 + TK), 0, 1);
            int rb = (tid & 127) >> 2, cb = (tid & 3) << 5;
            __builtin_prefetch(B + (size_t)(k0 + TK + rb) * Nc + bn + cb, 0, 1);
        }
        __syncthreads();

        // ---- fragments (ISA 7.12.2 16-bit layouts, wave32) ----
        Frag16 a[2], b[4];
#pragma unroll
        for (int mi = 0; mi < 2; ++mi) {
            const int arow = wm * 32 + mi * 16 + row16;
            // A 16x32: lanes 0-15: K[0..7]+K[16..23]; lanes 16-31: K[8..15]+K[24..31]
            a[mi].h[0] = *(const u16x8*)&As[arow][kh * 8];
            a[mi].h[1] = *(const u16x8*)&As[arow][16 + kh * 8];
        }
#pragma unroll
        for (int ni = 0; ni < 4; ++ni) {
            const int bcol = wn * 64 + ni * 16 + row16;
            // B 32x16: lanes 0-15: K[0..7]+K[16..23]; lanes 16-31: K[8..15]+K[24..31]
            b[ni].h[0] = *(const u16x8*)&Bs[bcol][kh * 8];
            b[ni].h[1] = *(const u16x8*)&Bs[bcol][16 + kh * 8];
        }
#pragma unroll
        for (int mi = 0; mi < 2; ++mi)
#pragma unroll
            for (int ni = 0; ni < 4; ++ni)
                acc[mi][ni] = __builtin_amdgcn_wmma_f32_16x16x32_bf16(
                    false, a[mi].bf, false, b[ni].bf, (short)0, acc[mi][ni],
                    false, false);
        __syncthreads();
    }

    // ---- store C (+bias): VGPR i -> M = (lane>>4)*8 + i, N = lane&15 ----
#pragma unroll
    for (int mi = 0; mi < 2; ++mi) {
        const int crow0 = bm + wm * 32 + mi * 16 + (lane >> 4) * 8;
#pragma unroll
        for (int ni = 0; ni < 4; ++ni) {
            const int ccol = bn + wn * 64 + ni * 16 + (lane & 15);
            const float bv = bias ? bias[ccol] : 0.0f;
#pragma unroll
            for (int i = 0; i < 8; ++i) {
                int r = crow0 + i;
                if (r < M) C[(size_t)r * Nc + ccol] = acc[mi][ni][i] + bv;
            }
        }
    }
}

// ---------------------------------------------------------------------------
// el[n,h] = feat[n,h,:]·al[h,:],  er[n,h] = feat[n,h,:]·ar[h,:]
// ---------------------------------------------------------------------------
__global__ void node_scores_kernel(const float* __restrict__ feat,
                                   const float* __restrict__ al,
                                   const float* __restrict__ ar,
                                   float* __restrict__ el, float* __restrict__ er,
                                   int Nn) {
    int t = blockIdx.x * blockDim.x + threadIdx.x;
    if (t >= Nn * H_) return;
    int n = t >> 2, h = t & 3;
    const float* f  = feat + (size_t)n * HID_ + h * D_;
    const float* a1 = al + h * D_;
    const float* a2 = ar + h * D_;
    float s1 = 0.f, s2 = 0.f;
#pragma unroll
    for (int d = 0; d < D_; d += 4) {
        float4 fv = *(const float4*)(f + d);
        float4 av = *(const float4*)(a1 + d);
        float4 bv = *(const float4*)(a2 + d);
        s1 += fv.x * av.x + fv.y * av.y + fv.z * av.z + fv.w * av.w;
        s2 += fv.x * bv.x + fv.y * bv.y + fv.z * bv.z + fv.w * bv.w;
    }
    el[t] = s1;
    er[t] = s2;
}

// wae[k,h] = sum_d We[k, h*D+d] * ae[h,d]
__global__ void compute_wae_kernel(const float* __restrict__ We,
                                   const float* __restrict__ ae,
                                   float* __restrict__ wae) {
    int t = blockIdx.x * blockDim.x + threadIdx.x;
    if (t >= HID_ * H_) return;
    int k = t >> 2, h = t & 3;
    float s = 0.f;
#pragma unroll 4
    for (int d = 0; d < D_; ++d)
        s += We[(size_t)k * HID_ + h * D_ + d] * ae[h * D_ + d];
    wae[t] = s;
}

// uae[j,h] = eenc_w[j,:]·wae[:,h] ; bae[h] = eenc_b·wae[:,h]
__global__ void compute_uae_kernel(const float* __restrict__ eenc_w,
                                   const float* __restrict__ eenc_b,
                                   const float* __restrict__ wae,
                                   float* __restrict__ uae, float* __restrict__ bae) {
    int t = blockIdx.x * blockDim.x + threadIdx.x;
    if (t < EIN_ * H_) {
        int j = t >> 2, h = t & 3;
        float s = 0.f;
        for (int k = 0; k < HID_; ++k)
            s += eenc_w[(size_t)j * HID_ + k] * wae[k * H_ + h];
        uae[t] = s;
    } else if (t < EIN_ * H_ + H_) {
        int h = t - EIN_ * H_;
        float s = 0.f;
        for (int k = 0; k < HID_; ++k)
            s += eenc_b[k] * wae[k * H_ + h];
        bae[h] = s;
    }
}

__global__ void init_layer_kernel(float* __restrict__ mmax, float* __restrict__ ssum,
                                  float* __restrict__ agg, int total, int nh) {
    int t = blockIdx.x * blockDim.x + threadIdx.x;
    if (t < total) agg[t] = 0.f;
    if (t < nh) { mmax[t] = -3.0e38f; ssum[t] = 0.f; }
}

// ---------------------------------------------------------------------------
// One wave per edge: ee = efeat·uae + bae (shuffle-reduced), then
// e = leaky_relu(el[src]+er[dst]+ee), store, atomic segment-max into m[dst].
// uae/bae staged to LDS via async global->LDS copy when available.
// ---------------------------------------------------------------------------
__global__ __launch_bounds__(256)
void edge_logit_kernel(const float* __restrict__ efeat,
                       const float* __restrict__ uae, const float* __restrict__ bae,
                       const float* __restrict__ el, const float* __restrict__ er,
                       const int* __restrict__ src, const int* __restrict__ dst,
                       float* __restrict__ elog, float* __restrict__ mmax, int Ne) {
    __shared__ float s_uae[EIN_ * H_];
    __shared__ float s_bae[H_];

    const int tid = threadIdx.x;
#if HAS_ASYNC_COPY
    // 256 threads: one dword each for uae, 4 threads for bae.
    // Builtin takes generic int* pointers on this toolchain.
    __builtin_amdgcn_global_load_async_to_lds_b32(
        (int*)(uae + tid), (int*)&s_uae[tid], 0, 0);
    if (tid < H_)
        __builtin_amdgcn_global_load_async_to_lds_b32(
            (int*)(bae + tid), (int*)&s_bae[tid], 0, 0);
    wait_async0();
#else
    s_uae[tid] = uae[tid];
    if (tid < H_) s_bae[tid] = bae[tid];
#endif
    __syncthreads();

    int wid  = (blockIdx.x * blockDim.x + tid) >> 5;
    int lane = tid & 31;
    if (wid >= Ne) return;

    const float* ef = efeat + (size_t)wid * EIN_;
    float v0 = ef[lane];
    float v1 = ef[lane + 32];
    float acc[H_];
#pragma unroll
    for (int h = 0; h < H_; ++h)
        acc[h] = v0 * s_uae[lane * H_ + h] + v1 * s_uae[(lane + 32) * H_ + h];
#pragma unroll
    for (int off = 16; off > 0; off >>= 1) {
#pragma unroll
        for (int h = 0; h < H_; ++h)
            acc[h] += __shfl_down(acc[h], off, 32);
    }
    if (lane == 0) {
        int s = src[wid], d = dst[wid];
#pragma unroll
        for (int h = 0; h < H_; ++h) {
            float e = el[s * H_ + h] + er[d * H_ + h] + acc[h] + s_bae[h];
            e = (e > 0.f) ? e : 0.2f * e;   // leaky_relu(0.2)
            elog[(size_t)wid * H_ + h] = e;
            atomicMaxFloat(&mmax[d * H_ + h], e);
        }
    }
}

__global__ void edge_exp_sum_kernel(const int* __restrict__ dst,
                                    const float* __restrict__ mmax,
                                    float* __restrict__ elog,  // in: logits, out: exp
                                    float* __restrict__ ssum, int Ne) {
    int t = blockIdx.x * blockDim.x + threadIdx.x;
    if (t >= Ne * H_) return;
    int e = t >> 2, h = t & 3;
    int d = dst[e];
    float a = __expf(elog[t] - mmax[d * H_ + h]);
    elog[t] = a;
    atomicAdd(&ssum[d * H_ + h], a);
}

// 64 threads per edge: agg[dst,c] += (a/ssum)[h(c)] * feat[src,c]
__global__ __launch_bounds__(256)
void aggregate_kernel(const float* __restrict__ a, const float* __restrict__ ssum,
                      const float* __restrict__ feat,
                      const int* __restrict__ src, const int* __restrict__ dst,
                      float* __restrict__ agg, int Ne) {
    int gt = blockIdx.x * blockDim.x + threadIdx.x;
    int e  = gt >> 6;
    if (e >= Ne) return;
    int c  = (gt & 63) << 2;   // 0..252, stays within one head (64-aligned blocks)
    int s  = src[e], d = dst[e];
    int h  = c >> 6;
    float alpha = a[(size_t)e * H_ + h] / ssum[d * H_ + h];
    float4 fv = *(const float4*)(feat + (size_t)s * HID_ + c);
    float* out = agg + (size_t)d * HID_ + c;
    atomicAdd(out + 0, alpha * fv.x);
    atomicAdd(out + 1, alpha * fv.y);
    atomicAdd(out + 2, alpha * fv.z);
    atomicAdd(out + 3, alpha * fv.w);
}

// One wave per node: x2 = LN(relu(agg) + x) * g + b  (wave32 shuffle reduce)
__global__ __launch_bounds__(256)
void relu_res_ln_kernel(const float* __restrict__ agg, const float* __restrict__ xin,
                        const float* __restrict__ g, const float* __restrict__ b,
                        float* __restrict__ xout, int Nn) {
    int w    = (blockIdx.x * blockDim.x + threadIdx.x) >> 5;
    int lane = threadIdx.x & 31;
    if (w >= Nn) return;
    const float* ag = agg + (size_t)w * HID_;
    const float* xi = xin + (size_t)w * HID_;
    float y[8];
    float sum = 0.f, sq = 0.f;
#pragma unroll
    for (int j = 0; j < 8; ++j) {
        int c = lane + j * 32;
        float v = ag[c];
        v = (v > 0.f) ? v : 0.f;
        v += xi[c];
        y[j] = v;
        sum += v;
        sq  += v * v;
    }
#pragma unroll
    for (int off = 16; off > 0; off >>= 1) {
        sum += __shfl_xor(sum, off, 32);
        sq  += __shfl_xor(sq, off, 32);
    }
    float mu  = sum * (1.0f / HID_);
    float var = sq * (1.0f / HID_) - mu * mu;
    float inv = rsqrtf(var + 1e-5f);
    float* xo = xout + (size_t)w * HID_;
#pragma unroll
    for (int j = 0; j < 8; ++j) {
        int c = lane + j * 32;
        xo[c] = (y[j] - mu) * inv * g[c] + b[c];
    }
}

// ---------------------------------------------------------------------------
extern "C" void kernel_launch(void* const* d_in, const int* in_sizes, int n_in,
                              void* d_out, int out_size, void* d_ws, size_t ws_size,
                              hipStream_t stream) {
    const float* h      = (const float*)d_in[0];
    const float* efeat  = (const float*)d_in[1];
    const int*   src    = (const int*)d_in[2];
    const int*   dst    = (const int*)d_in[3];
    const float* enc_w  = (const float*)d_in[4];
    const float* enc_b  = (const float*)d_in[5];
    const float* eenc_w = (const float*)d_in[6];
    const float* eenc_b = (const float*)d_in[7];
    const float* Wn     = (const float*)d_in[8];
    const float* We     = (const float*)d_in[9];
    const float* al     = (const float*)d_in[10];
    const float* ar     = (const float*)d_in[11];
    const float* ae     = (const float*)d_in[12];
    const float* ln_g   = (const float*)d_in[13];
    const float* ln_b   = (const float*)d_in[14];
    const float* out_w  = (const float*)d_in[15];
    const float* out_b  = (const float*)d_in[16];

    const int Nn = in_sizes[0] / IN_;
    const int Ne = in_sizes[2];

    const size_t Nf = (size_t)Nn * HID_;
    const size_t NH = (size_t)Nn * H_;

    float* ws   = (float*)d_ws;
    float* x    = ws;
    float* x2   = x + Nf;
    float* feat = x2 + Nf;
    float* agg  = feat + Nf;
    float* el   = agg + Nf;
    float* er   = el + NH;
    float* mmax = er + NH;
    float* ssum = mmax + NH;
    float* elog = ssum + NH;
    float* wae  = elog + (size_t)Ne * H_;
    float* uae  = wae + HID_ * H_;
    float* bae  = uae + EIN_ * H_;
    (void)ws_size; (void)n_in; (void)out_size;

    const dim3 blk(256);
    const dim3 gemm_grid(HID_ / TN, (Nn + TM - 1) / TM);

    // node encoder: x = h @ enc_w + enc_b
    gemm_bias_wmma<<<gemm_grid, blk, 0, stream>>>(h, enc_w, enc_b, x, Nn, IN_, HID_);

    for (int l = 0; l < L_; ++l) {
        const float* Wnl = Wn + (size_t)l * HID_ * HID_;
        const float* Wel = We + (size_t)l * HID_ * HID_;
        const float* all = al + (size_t)l * H_ * D_;
        const float* arl = ar + (size_t)l * H_ * D_;
        const float* ael = ae + (size_t)l * H_ * D_;

        // feat = x @ Wn[l]
        gemm_bias_wmma<<<gemm_grid, blk, 0, stream>>>(x, Wnl, nullptr, feat, Nn, HID_, HID_);

        node_scores_kernel<<<(Nn * H_ + 255) / 256, blk, 0, stream>>>(feat, all, arl, el, er, Nn);

        compute_wae_kernel<<<(HID_ * H_ + 255) / 256, blk, 0, stream>>>(Wel, ael, wae);
        compute_uae_kernel<<<(EIN_ * H_ + H_ + 255) / 256, blk, 0, stream>>>(eenc_w, eenc_b, wae, uae, bae);

        init_layer_kernel<<<((int)Nf + 255) / 256, blk, 0, stream>>>(mmax, ssum, agg, (int)Nf, (int)NH);

        edge_logit_kernel<<<(Ne + 7) / 8, blk, 0, stream>>>(efeat, uae, bae, el, er,
                                                            src, dst, elog, mmax, Ne);

        edge_exp_sum_kernel<<<(Ne * H_ + 255) / 256, blk, 0, stream>>>(dst, mmax, elog, ssum, Ne);

        aggregate_kernel<<<(Ne * 64 + 255) / 256, blk, 0, stream>>>(elog, ssum, feat,
                                                                    src, dst, agg, Ne);

        relu_res_ln_kernel<<<(Nn * 32 + 255) / 256, blk, 0, stream>>>(
            agg, x, ln_g + (size_t)l * HID_, ln_b + (size_t)l * HID_, x2, Nn);

        float* tmp = x; x = x2; x2 = tmp;
    }

    // out = x @ out_w + out_b
    gemm_bias_wmma<<<dim3(OUT_ / TN, (Nn + TM - 1) / TM), blk, 0, stream>>>(
        x, out_w, out_b, (float*)d_out, Nn, HID_, OUT_);
}
